// LmLSTM_58248346468592
// MI455X (gfx1250) — compile-verified, matrix operand order checked by reading, not run on hardware
//
#include <hip/hip_runtime.h>
#include <math.h>

#define B_ 8
#define T_ 512
#define H_ 1024
#define V_ 32000
#define L_ 2
#define M_ (B_ * T_)   // 4096 rows (b*T + t)
#define G_ (4 * H_)    // 4096 gate width

typedef __bf16 bf16;
typedef __attribute__((ext_vector_type(8)))  __bf16 v8bf;
typedef __attribute__((ext_vector_type(16))) __bf16 v16bf;
typedef __attribute__((ext_vector_type(8)))  float  v8f;
typedef __attribute__((ext_vector_type(4)))  int    v4i;

typedef __attribute__((address_space(1))) v4i as1_v4i;
typedef __attribute__((address_space(3))) v4i as3_v4i;

__device__ __forceinline__ bf16 f2bf(float f) { return (bf16)f; }   // native v_cvt, RNE

__device__ __forceinline__ v16bf mk16(v8bf lo, v8bf hi) {
    v16bf r;
#pragma unroll
    for (int i = 0; i < 8; ++i) { r[i] = lo[i]; r[i + 8] = hi[i]; }
    return r;
}

__device__ __forceinline__ float sigmoidf_(float x) { return 1.0f / (1.0f + __expf(-x)); }

// ---- async memory->LDS staging (gfx1250 GLOBAL_LOAD_ASYNC_TO_LDS_B128, ASYNCcnt) ----
#if __has_builtin(__builtin_amdgcn_global_load_async_to_lds_b128)
#define ASYNC_LDS 1
#else
#define ASYNC_LDS 0
#endif

__device__ __forceinline__ void stage16B(const bf16* __restrict__ g, bf16* l) {
#if ASYNC_LDS
    // generic->AS1 / AS3 via integer cast (LDS offset = low 32 bits of generic ptr)
    __builtin_amdgcn_global_load_async_to_lds_b128(
        (as1_v4i*)(unsigned long long)(const void*)g,
        (as3_v4i*)(unsigned int)(unsigned long long)(void*)l,
        0, 0);
#else
    *(v8bf*)l = *(const v8bf*)g;
#endif
}

__device__ __forceinline__ void wait_async_lds() {
#if ASYNC_LDS
#if __has_builtin(__builtin_amdgcn_s_wait_asynccnt)
    __builtin_amdgcn_s_wait_asynccnt(0);
#else
    asm volatile("s_wait_asynccnt 0x0" ::: "memory");
#endif
#endif
}

// ---------------- embedding gather -> bf16 activations ----------------
__global__ __launch_bounds__(256) void gather_embed_bf16(
    const int* __restrict__ tok, const float* __restrict__ emb, bf16* __restrict__ x)
{
    size_t idx = (size_t)blockIdx.x * 256 + threadIdx.x;   // over M_*H_
    int m = (int)(idx >> 10);
    int h = (int)(idx & (H_ - 1));
    x[idx] = f2bf(emb[(size_t)tok[m] * H_ + h]);
}

// ---------------- Wh (H x 4H, f32) -> WhT (4H x H, bf16, K-contiguous) ----------------
__global__ __launch_bounds__(256) void transpose_conv_bf16(
    const float* __restrict__ W, bf16* __restrict__ WT)
{
    size_t idx = (size_t)blockIdx.x * 256 + threadIdx.x;   // over G_*H_
    int n = (int)(idx >> 10);          // gate column 0..4095
    int k = (int)(idx & (H_ - 1));     // k 0..1023
    WT[idx] = f2bf(W[(size_t)k * G_ + n]);
}

__global__ __launch_bounds__(256) void zero_f32(float* __restrict__ p, int n)
{
    int i = blockIdx.x * 256 + threadIdx.x;
    if (i < n) p[i] = 0.0f;
}

// ---------------- big WMMA GEMM: C[M,N] = bf16(A[M,K]) * bf16(Bw[K,N]) + bias[N] --------
// 256 threads = 8 waves; WG tile 128x128, wave tile 64x32, K-chunk 32, double-buffered LDS.
// A (bf16 row-major) staged via async-to-LDS; B (f32 row-major) converted to bf16 and
// transposed to [n][k] while staged so fragment loads are clean K-contiguous b128s.
__global__ __launch_bounds__(256) void wmma_gemm_bf16f32(
    const bf16* __restrict__ A, const float* __restrict__ Bw,
    const float* __restrict__ bias, float* __restrict__ C,
    int M, int N, int K)
{
    __shared__ bf16 As[2][128 * 40];   // [m][k], stride 40 kills bank conflicts
    __shared__ bf16 Bs[2][128 * 40];   // transposed [n][k]

    const int tid  = threadIdx.x;
    const int wgM  = blockIdx.y * 128;
    const int wgN  = blockIdx.x * 128;
    const int w    = tid >> 5;            // wave32
    const int lane = tid & 31;
    const int wm   = (w >> 2) * 64;
    const int wn   = (w & 3) * 32;
    const int lm   = lane & 15;
    const int kh   = (lane >> 4) * 8;     // ISA A/B 16-bit fragment K-half select

    const int arow = tid >> 1, acol = (tid & 1) * 16;
    const int brow = tid >> 3, bcol = (tid & 7) * 16;

    v8f acc[4][2] = {};

    const int nk = K / 32;

    // ---- tile stagers ----
    auto stageA = [&](int kb, int buf) {
        const bf16* ga = &A[(size_t)(wgM + arow) * K + kb * 32 + acol];
        stage16B(ga,     &As[buf][arow * 40 + acol]);
        stage16B(ga + 8, &As[buf][arow * 40 + acol + 8]);
    };
    auto stageB = [&](int kb, int buf) {
        const float* gb = &Bw[(size_t)(kb * 32 + brow) * N + wgN + bcol];
        float4 f0 = *(const float4*)(gb + 0);
        float4 f1 = *(const float4*)(gb + 4);
        float4 f2 = *(const float4*)(gb + 8);
        float4 f3 = *(const float4*)(gb + 12);
        float fv[16] = { f0.x, f0.y, f0.z, f0.w, f1.x, f1.y, f1.z, f1.w,
                         f2.x, f2.y, f2.z, f2.w, f3.x, f3.y, f3.z, f3.w };
#pragma unroll
        for (int i = 0; i < 16; ++i)
            Bs[buf][(bcol + i) * 40 + brow] = f2bf(fv[i]);
    };

    // prologue: fill buffer 0
    stageA(0, 0);
    stageB(0, 0);
    wait_async_lds();
    __syncthreads();

    for (int kb = 0; kb < nk; ++kb) {
        const int cur = kb & 1, nxt = cur ^ 1;
        if (kb + 1 < nk) {        // overlap next tile fill with this tile's WMMAs
            stageA(kb + 1, nxt);
            __builtin_prefetch(&Bw[(size_t)((kb + 1) * 32 + brow) * N + wgN + bcol], 0, 1);
            stageB(kb + 1, nxt);
        }

        v16bf af[4], bfr[2];
#pragma unroll
        for (int i = 0; i < 4; ++i) {
            int r = wm + 16 * i + lm;
            af[i] = mk16(*(const v8bf*)&As[cur][r * 40 + kh],
                         *(const v8bf*)&As[cur][r * 40 + 16 + kh]);
        }
#pragma unroll
        for (int j = 0; j < 2; ++j) {
            int n = wn + 16 * j + lm;
            bfr[j] = mk16(*(const v8bf*)&Bs[cur][n * 40 + kh],
                          *(const v8bf*)&Bs[cur][n * 40 + 16 + kh]);
        }
#pragma unroll
        for (int i = 0; i < 4; ++i)
#pragma unroll
            for (int j = 0; j < 2; ++j)
                acc[i][j] = __builtin_amdgcn_wmma_f32_16x16x32_bf16(
                    false, af[i], false, bfr[j], (short)0, acc[i][j], false, false);

        wait_async_lds();         // next buffer's async fills landed
        __syncthreads();          // everyone done reading cur / staging nxt
    }

    // C/D layout: lanes 0-15 M=r, lanes 16-31 M=r+8; N = lane&15
    const int moff = (lane >> 4) * 8;
#pragma unroll
    for (int i = 0; i < 4; ++i)
#pragma unroll
        for (int j = 0; j < 2; ++j) {
            int gn = wgN + wn + 16 * j + lm;
            float bs = bias[gn];
#pragma unroll
            for (int r = 0; r < 8; ++r) {
                int gm = wgM + wm + 16 * i + r + moff;
                C[(size_t)gm * N + gn] = acc[i][j][r] + bs;
            }
        }
}

// ---------------- one LSTM time step ----------------
// 16 WGs x 128 threads (4 waves). Each wave owns one 16-wide h-column tile and
// holds 4 WMMA accumulators (i,f,g,o) over the four Wh column strips, so the
// gate nonlinearity is pure per-lane math on the WMMA C layout.
__global__ __launch_bounds__(128) void lstm_step(
    const float* __restrict__ xW, const bf16* __restrict__ WhT,
    const float* __restrict__ h_in, float* __restrict__ h_out,
    float* __restrict__ cst, const int* __restrict__ tok,
    bf16* __restrict__ y, int t)
{
    __shared__ bf16 hs[16 * 1032];   // h_prev, rows padded to 16 (rows 8..15 zero)

    const int tid = threadIdx.x;
    for (int i = tid; i < 16 * H_; i += 128) {
        int r = i >> 10, c = i & (H_ - 1);
        hs[r * 1032 + c] = f2bf(h_in[i]);
    }
    __syncthreads();

    const int w    = tid >> 5;
    const int lane = tid & 31;
    const int j0   = (blockIdx.x * 4 + w) * 16;   // owned h columns [j0, j0+16)
    const int lm   = lane & 15;
    const int kh   = (lane >> 4) * 8;

    v8f acc[4] = {};
    for (int kc = 0; kc < H_; kc += 32) {
        v16bf a = mk16(*(const v8bf*)&hs[lm * 1032 + kc + kh],
                       *(const v8bf*)&hs[lm * 1032 + kc + 16 + kh]);
#pragma unroll
        for (int g = 0; g < 4; ++g) {
            const bf16* wp = &WhT[(size_t)((g << 10) + j0 + lm) * H_ + kc + kh];
            v16bf bv = mk16(*(const v8bf*)wp, *(const v8bf*)(wp + 16));
            acc[g] = __builtin_amdgcn_wmma_f32_16x16x32_bf16(
                false, a, false, bv, (short)0, acc[g], false, false);
        }
    }

    if (lane < 16) {   // valid batches live in M rows 0..7 (lanes 0-15, vgpr r<8)
        int col = j0 + lm;
#pragma unroll
        for (int r = 0; r < 8; ++r) {
            int b   = r;
            int row = b * T_ + t;
            const float* z = &xW[(size_t)row * G_];
            float zi = acc[0][r] + z[col];
            float zf = acc[1][r] + z[1024 + col];
            float zg = acc[2][r] + z[2048 + col];
            float zo = acc[3][r] + z[3072 + col];
            float iv = sigmoidf_(zi), fv = sigmoidf_(zf);
            float gv = tanhf(zg),     ov = sigmoidf_(zo);
            float cold = cst[b * H_ + col];
            float cn = fv * cold + iv * gv;
            float hn = ov * tanhf(cn);
            bool keep = tok[row] != 0;
            float hold = h_in[b * H_ + col];
            cn = keep ? cn : cold;
            hn = keep ? hn : hold;
            cst[b * H_ + col]   = cn;
            h_out[b * H_ + col] = hn;
            y[(size_t)row * H_ + col] = f2bf(hn);
        }
    }
}

// ---------------- host-side orchestration ----------------
extern "C" void kernel_launch(void* const* d_in, const int* in_sizes, int n_in,
                              void* d_out, int out_size, void* d_ws, size_t ws_size,
                              hipStream_t stream)
{
    (void)in_sizes; (void)n_in; (void)out_size; (void)ws_size;
    const int*   tokens = (const int*)  d_in[0];
    const float* embed  = (const float*)d_in[1];
    const float* Wi     = (const float*)d_in[2];
    const float* Wh     = (const float*)d_in[3];
    const float* bias   = (const float*)d_in[4];
    const float* Wo     = (const float*)d_in[5];
    const float* bo     = (const float*)d_in[6];
    float* out = (float*)d_out;

    char* ws = (char*)d_ws;
    size_t off = 0;
    auto take = [&](size_t bytes) -> char* {
        char* p = ws + off;
        off += (bytes + 255) & ~(size_t)255;
        return p;
    };
    bf16*  xA  = (bf16*) take((size_t)M_ * H_ * 2);   // 8 MB
    bf16*  xB  = (bf16*) take((size_t)M_ * H_ * 2);   // 8 MB
    float* xW  = (float*)take((size_t)M_ * G_ * 4);   // 64 MB
    bf16*  WhT = (bf16*) take((size_t)G_ * H_ * 2);   // 8 MB
    float* hA  = (float*)take((size_t)16 * H_ * 4);
    float* hB  = (float*)take((size_t)16 * H_ * 4);
    float* cS  = (float*)take((size_t)16 * H_ * 4);

    gather_embed_bf16<<<(M_ * H_) / 256, 256, 0, stream>>>(tokens, embed, xA);

    bf16* cur = xA;
    bf16* nxt = xB;
    for (int l = 0; l < L_; ++l) {
        transpose_conv_bf16<<<(G_ * H_) / 256, 256, 0, stream>>>(
            Wh + (size_t)l * H_ * G_, WhT);
        wmma_gemm_bf16f32<<<dim3(G_ / 128, M_ / 128), 256, 0, stream>>>(
            cur, Wi + (size_t)l * H_ * G_, bias + (size_t)l * G_, xW, M_, G_, H_);
        zero_f32<<<(16 * H_) / 256, 256, 0, stream>>>(hA, 16 * H_);
        zero_f32<<<(16 * H_) / 256, 256, 0, stream>>>(hB, 16 * H_);
        zero_f32<<<(16 * H_) / 256, 256, 0, stream>>>(cS, 16 * H_);
        for (int t = 0; t < T_; ++t) {
            const float* hin  = (t & 1) ? hB : hA;
            float*       hout = (t & 1) ? hA : hB;
            lstm_step<<<16, 128, 0, stream>>>(xW, WhT, hin, hout, cS, tokens, nxt, t);
        }
        bf16* tmp = cur; cur = nxt; nxt = tmp;
    }

    wmma_gemm_bf16f32<<<dim3(V_ / 128, M_ / 128), 256, 0, stream>>>(
        cur, Wo, bo, out, M_, V_, H_);
}
